// Model_14998025797662
// MI455X (gfx1250) — matrix operand order repaired and checked by reading, not compiled, not run on hardware
//
#include <hip/hip_runtime.h>
#include <hip/hip_bf16.h>
#include <math.h>

// ---------------- problem constants (from reference) ----------------
#define BB   16
#define LL   1024
#define CIN  12
#define DMOD 256
#define DI   512
#define DS   16
#define DCV  4
#define DTR  16
#define NC   10
#define NH   8
#define EK   3
#define BL   (BB * LL)        // 16384 rows

typedef _Float16 v16h __attribute__((ext_vector_type(16)));
typedef _Float16 v8h  __attribute__((ext_vector_type(8)));
typedef float    v8f  __attribute__((ext_vector_type(8)));

// ---------------- workspace layout (bytes) ----------------
#define OFF_EMB   0ull                                   // BL*DM f16   = 8 MB
#define OFF_WIN   (OFF_EMB  + (size_t)BL*DMOD*2)         // 1024*256 f16
#define OFF_XZ    (OFF_WIN  + (size_t)1024*DMOD*2)       // BL*1024 f16 = 32 MB
#define OFF_U     (OFF_XZ   + (size_t)BL*1024*2)         // BL*512 f16
#define OFF_WX    (OFF_U    + (size_t)BL*DI*2)           // 48*512 f16
#define OFF_XDBL  (OFF_WX   + (size_t)48*DI*2)           // BL*48 f32
#define OFF_DT    (OFF_XDBL + (size_t)BL*48*4)           // BL*512 f16
#define OFF_YG    (OFF_DT   + (size_t)BL*DI*2)           // BL*512 f16
#define OFF_WOUT  (OFF_YG   + (size_t)BL*DI*2)           // 256*512 f16
#define OFF_MO    (OFF_WOUT + (size_t)DMOD*DI*2)         // BL*256 f32
#define OFF_MA    (OFF_MO   + (size_t)BL*DMOD*4)         // BL*256 f32
#define OFF_SB    (OFF_MA   + (size_t)BL*DMOD*4)         // BL f32
#define OFF_LG    (OFF_SB   + (size_t)BL*4)              // BL*NC f32
#define WS_NEED   (OFF_LG   + (size_t)BL*NC*4)           // ~125 MB

// ---------------- f32 -> f16 weight convert ----------------
__global__ void k_f32_to_f16(const float* __restrict__ in, _Float16* __restrict__ out, int n) {
    int i = blockIdx.x * blockDim.x + threadIdx.x;
    if (i < n) out[i] = (_Float16)in[i];
}

// ---------------- token conv (k=3, replicate pad) + sinusoidal PE ----------------
__global__ void k_embed(const float* __restrict__ x_enc, const float* __restrict__ wconv,
                        _Float16* __restrict__ emb) {
    int row = blockIdx.x;               // b*L + l
    int b = row >> 10, l = row & (LL - 1);
    int d = threadIdx.x;
    float s = 0.f;
    #pragma unroll
    for (int k = 0; k < EK; ++k) {
        int ls = l + k - 1;
        ls = ls < 0 ? 0 : (ls > LL - 1 ? LL - 1 : ls);
        const float* xp = x_enc + ((size_t)b * LL + ls) * CIN;
        #pragma unroll
        for (int c = 0; c < CIN; ++c)
            s += xp[c] * wconv[(d * CIN + c) * EK + k];
    }
    int i2 = d & ~1;
    float div = __expf(-(float)i2 * (9.210340371976184f / 256.0f));
    float arg = (float)l * div;
    s += (d & 1) ? __cosf(arg) : __sinf(arg);
    emb[(size_t)row * DMOD + d] = (_Float16)s;
}

// ------------- fragment load helpers (cdna5 §7.12.2 layouts, wave32) -------------
__device__ __forceinline__ v16h load_a_frag(const _Float16* __restrict__ A,
                                            int arow, int K, int k0, int hi) {
    const _Float16* ap = A + (size_t)arow * K + k0 + hi * 8;
    v8h lo = *(const v8h*)ap;
    v8h hv = *(const v8h*)(ap + 16);
    v16h f;
    #pragma unroll
    for (int i = 0; i < 8; ++i) { f[i] = lo[i]; f[i + 8] = hv[i]; }
    return f;
}
__device__ __forceinline__ v16h load_b_frag(const _Float16* __restrict__ W,
                                            int nrow, int K, int k0, int hi) {
    const _Float16* wp = W + (size_t)nrow * K + k0 + hi * 8;
    v8h lo = *(const v8h*)wp;
    v8h hv = *(const v8h*)(wp + 16);
    v16h f;
    #pragma unroll
    for (int i = 0; i < 8; ++i) { f[i] = lo[i]; f[i + 8] = hv[i]; }
    return f;
}

// ------------- WMMA GEMM, MT x NT 16x16-tiles per wave ---------------------------
// C[M,N] = A[M,K] @ W[N,K]^T, f16 operands, f32 accumulate.
// __launch_bounds__(...,1): trade occupancy for VGPRs -- accumulators must not spill;
// load latency is hidden by the compiler's software pipelining, not by wave count.
template <int MT, int NT, bool OUT_F16>
__launch_bounds__(128, 1)
__global__ void k_gemm_nt_wmma_blk(const _Float16* __restrict__ A,
                                   const _Float16* __restrict__ W,
                                   void* __restrict__ Cout,
                                   int M, int N, int K) {
    const int lane = threadIdx.x & 31;
    const int wave = threadIdx.x >> 5;
    const int tN = N / (16 * NT);
    const int total = (M / (16 * MT)) * tN;
    int t = blockIdx.x * (blockDim.x >> 5) + wave;
    if (t >= total) return;
    const int tm = t / tN, tn = t % tN;
    const int m = lane & 15, hi = lane >> 4;
    const int r0 = tm * (16 * MT), c0 = tn * (16 * NT);

    v8f acc[MT * NT];
    v8f zero = {};
    #pragma unroll
    for (int i = 0; i < MT * NT; ++i) acc[i] = zero;

    for (int k0 = 0; k0 < K; k0 += 32) {
        __builtin_prefetch(A + (size_t)(r0 + m) * K + k0 + 64, 0, 3);  // near-cache
        v16h af[MT];
        #pragma unroll
        for (int i = 0; i < MT; ++i) af[i] = load_a_frag(A, r0 + i * 16 + m, K, k0, hi);
        v16h bf[NT];
        #pragma unroll
        for (int j = 0; j < NT; ++j) bf[j] = load_b_frag(W, c0 + j * 16 + m, K, k0, hi);
        #pragma unroll
        for (int i = 0; i < MT; ++i)
            #pragma unroll
            for (int j = 0; j < NT; ++j)
                acc[i * NT + j] = __builtin_amdgcn_wmma_f32_16x16x32_f16(
                    false, af[i], false, bf[j], (short)0, acc[i * NT + j], false, false);
    }
    #pragma unroll
    for (int i = 0; i < MT; ++i) {
        #pragma unroll
        for (int j = 0; j < NT; ++j) {
            const int rb = r0 + i * 16 + hi * 8;
            const int col = c0 + j * 16 + m;
            if (OUT_F16) {
                _Float16* C = (_Float16*)Cout;
                #pragma unroll
                for (int r = 0; r < 8; ++r)
                    C[(size_t)(rb + r) * N + col] = (_Float16)acc[i * NT + j][r];
            } else {
                float* C = (float*)Cout;
                #pragma unroll
                for (int r = 0; r < 8; ++r)
                    C[(size_t)(rb + r) * N + col] = acc[i * NT + j][r];
            }
        }
    }
}

// ---------------- causal depthwise conv (k=4) + bias + SiLU -> u (f16) ----------
__global__ void k_dconv_silu(const _Float16* __restrict__ xz, const float* __restrict__ dw,
                             const float* __restrict__ db, _Float16* __restrict__ u) {
    int row = blockIdx.x;
    int b = row >> 10, l = row & (LL - 1);
    int d = threadIdx.x;
    float s = db[d];
    #pragma unroll
    for (int k = 0; k < DCV; ++k) {
        int ls = l + k - (DCV - 1);
        if (ls >= 0)
            s += (float)xz[((size_t)(b * LL + ls)) * (2 * DI) + d] * dw[d * DCV + k];
    }
    float sig = 1.f / (1.f + __expf(-s));
    u[(size_t)row * DI + d] = (_Float16)(s * sig);
}

// ---------------- dt = softplus(dt_low @ Wdt^T + b) (f16 out) ----------------
__global__ void k_dt_softplus(const float* __restrict__ xdbl, const float* __restrict__ wdt,
                              const float* __restrict__ bdt, _Float16* __restrict__ dtb) {
    int row = blockIdx.x;
    int d = threadIdx.x;                  // 0..511
    const float* xp = xdbl + (size_t)row * 48;
    float s = bdt[d];
    #pragma unroll
    for (int j = 0; j < DTR; ++j) s += xp[j] * wdt[d * DTR + j];
    float sp = (s > 20.f) ? s : log1pf(__expf(s));
    dtb[(size_t)row * DI + d] = (_Float16)sp;
}

// ---------------- selective scan, one wave per (b, 32-channel group) --------------
// fused +u*D, *silu(z), f16 store. B/C broadcast via lane shuffles: no LDS, no sync.
__global__ void k_scan(const _Float16* __restrict__ dtb, const _Float16* __restrict__ ub,
                       const _Float16* __restrict__ xz, const float* __restrict__ xdbl,
                       const float* __restrict__ A_log, const float* __restrict__ Dvec,
                       _Float16* __restrict__ yg) {
    int b = blockIdx.x;                   // 0..15
    int lane = threadIdx.x;               // 0..31
    int d = blockIdx.y * 32 + lane;       // channel
    float a[DS], h[DS];
    #pragma unroll
    for (int s = 0; s < DS; ++s) { a[s] = -__expf(A_log[d * DS + s]); h[s] = 0.f; }
    float Dd = Dvec[d];
    for (int t = 0; t < LL; ++t) {
        int row = b * LL + t;
        // lane s holds B_s (s<16) / C_{s-16} (s>=16); broadcast via shuffle
        float bcv = xdbl[(size_t)row * 48 + DTR + lane];
        float dt = (float)dtb[(size_t)row * DI + d];
        float uu = (float)ub[(size_t)row * DI + d];
        float du = dt * uu;
        float y = 0.f;
        #pragma unroll
        for (int s = 0; s < DS; ++s) {
            float Bs = __shfl(bcv, s, 32);
            float Cs = __shfl(bcv, s + 16, 32);
            h[s] = __expf(dt * a[s]) * h[s] + du * Bs;
            y += h[s] * Cs;
        }
        float z = (float)xz[(size_t)row * (2 * DI) + DI + d];
        float sz = z / (1.f + __expf(-z));
        yg[(size_t)row * DI + d] = (_Float16)((y + uu * Dd) * sz);
    }
}

// ---------------- LayerNorm + SiLU ----------------
__global__ void k_ln_silu(const float* __restrict__ mo, const float* __restrict__ g,
                          const float* __restrict__ be, float* __restrict__ ma) {
    int row = blockIdx.x;
    int d = threadIdx.x;                  // 0..255
    __shared__ float red[DMOD];
    float x = mo[(size_t)row * DMOD + d];
    red[d] = x; __syncthreads();
    for (int s = 128; s; s >>= 1) { if (d < s) red[d] += red[d + s]; __syncthreads(); }
    float mu = red[0] * (1.f / DMOD);
    __syncthreads();
    float c = x - mu;
    red[d] = c * c; __syncthreads();
    for (int s = 128; s; s >>= 1) { if (d < s) red[d] += red[d + s]; __syncthreads(); }
    float var = red[0] * (1.f / DMOD);
    float v = c * rsqrtf(var + 1e-5f) * g[d] + be[d];
    ma[(size_t)row * DMOD + d] = v / (1.f + __expf(-v));
}

// ---------------- head stage 1: per (b,l): 10 class logits + max-head attn --------
__global__ void k_head1(const float* __restrict__ ma, const float* __restrict__ cls_w,
                        const float* __restrict__ attn_w, const float* __restrict__ attn_b,
                        const float* __restrict__ xmark,
                        float* __restrict__ sbuf, float* __restrict__ logitbuf) {
    int lane = threadIdx.x & 31, wave = threadIdx.x >> 5;
    int row = blockIdx.x * (blockDim.x >> 5) + wave;
    if (row >= BL) return;
    float acc[NC + NH];
    #pragma unroll
    for (int o = 0; o < NC + NH; ++o) acc[o] = 0.f;
    for (int d = lane; d < DMOD; d += 32) {
        float x = ma[(size_t)row * DMOD + d];
        #pragma unroll
        for (int nc = 0; nc < NC; ++nc) acc[nc] += x * cls_w[nc * DMOD + d];
        #pragma unroll
        for (int hh = 0; hh < NH; ++hh) acc[NC + hh] += x * attn_w[hh * DMOD + d];
    }
    #pragma unroll
    for (int off = 16; off; off >>= 1) {
        #pragma unroll
        for (int o = 0; o < NC + NH; ++o) acc[o] += __shfl_xor(acc[o], off, 32);
    }
    if (lane == 0) {
        float xm = xmark[row];
        #pragma unroll
        for (int nc = 0; nc < NC; ++nc) logitbuf[(size_t)row * NC + nc] = acc[nc] * xm;
        float mx = -3.0e38f;
        #pragma unroll
        for (int hh = 0; hh < NH; ++hh) mx = fmaxf(mx, acc[NC + hh] + attn_b[hh]);
        sbuf[row] = mx;
    }
}

// ---------------- head stage 2: softmax over L, weighted sum -> out[B,NC] ---------
__global__ void k_head2(const float* __restrict__ sbuf, const float* __restrict__ logitbuf,
                        float* __restrict__ out) {
    int b = blockIdx.x;
    int tid = threadIdx.x;                // 0..255
    __shared__ float red[256];
    float mx = -3.0e38f;
    for (int l = tid; l < LL; l += 256) mx = fmaxf(mx, sbuf[b * LL + l]);
    red[tid] = mx; __syncthreads();
    for (int s = 128; s; s >>= 1) { if (tid < s) red[tid] = fmaxf(red[tid], red[tid + s]); __syncthreads(); }
    mx = red[0]; __syncthreads();
    float se = 0.f;
    for (int l = tid; l < LL; l += 256) se += __expf(sbuf[b * LL + l] - mx);
    red[tid] = se; __syncthreads();
    for (int s = 128; s; s >>= 1) { if (tid < s) red[tid] += red[tid + s]; __syncthreads(); }
    float inv = 1.f / red[0]; __syncthreads();
    for (int nc = 0; nc < NC; ++nc) {
        float acc = 0.f;
        for (int l = tid; l < LL; l += 256)
            acc += logitbuf[(size_t)(b * LL + l) * NC + nc] * __expf(sbuf[b * LL + l] - mx);
        red[tid] = acc; __syncthreads();
        for (int s = 128; s; s >>= 1) { if (tid < s) red[tid] += red[tid + s]; __syncthreads(); }
        if (tid == 0) out[b * NC + nc] = red[0] * inv;
        __syncthreads();
    }
}

// ---------------- launcher ----------------
extern "C" void kernel_launch(void* const* d_in, const int* in_sizes, int n_in,
                              void* d_out, int out_size, void* d_ws, size_t ws_size,
                              hipStream_t stream) {
    (void)in_sizes; (void)n_in; (void)out_size;
    if (ws_size < WS_NEED) return;   // workspace too small (deterministic no-op)

    const float* x_enc     = (const float*)d_in[0];
    const float* x_mark    = (const float*)d_in[1];
    const float* tok_conv  = (const float*)d_in[2];
    const float* in_proj_w = (const float*)d_in[3];
    const float* dconv_w   = (const float*)d_in[4];
    const float* dconv_b   = (const float*)d_in[5];
    const float* x_proj_w  = (const float*)d_in[6];
    const float* dt_proj_w = (const float*)d_in[7];
    const float* dt_proj_b = (const float*)d_in[8];
    const float* A_log     = (const float*)d_in[9];
    const float* Dvec      = (const float*)d_in[10];
    const float* out_proj_w= (const float*)d_in[11];
    const float* ln_g      = (const float*)d_in[12];
    const float* ln_b      = (const float*)d_in[13];
    const float* cls_w     = (const float*)d_in[14];
    const float* attn_w    = (const float*)d_in[15];
    const float* attn_b    = (const float*)d_in[16];
    float* out = (float*)d_out;

    char* ws = (char*)d_ws;
    _Float16* emb   = (_Float16*)(ws + OFF_EMB);
    _Float16* w_in  = (_Float16*)(ws + OFF_WIN);
    _Float16* xz    = (_Float16*)(ws + OFF_XZ);
    _Float16* u     = (_Float16*)(ws + OFF_U);
    _Float16* w_x   = (_Float16*)(ws + OFF_WX);
    float*    xdbl  = (float*)   (ws + OFF_XDBL);
    _Float16* dtb   = (_Float16*)(ws + OFF_DT);
    _Float16* yg    = (_Float16*)(ws + OFF_YG);
    _Float16* w_out = (_Float16*)(ws + OFF_WOUT);
    float*    mo    = (float*)   (ws + OFF_MO);
    float*    ma    = (float*)   (ws + OFF_MA);
    float*    sbuf  = (float*)   (ws + OFF_SB);
    float*    lgbuf = (float*)   (ws + OFF_LG);

    // weight converts (f32 -> f16, keep original [N,K] layout)
    {
        int n1 = 1024 * DMOD;      // in_proj_w
        k_f32_to_f16<<<(n1 + 255) / 256, 256, 0, stream>>>(in_proj_w, w_in, n1);
        int n2 = 48 * DI;          // x_proj_w
        k_f32_to_f16<<<(n2 + 255) / 256, 256, 0, stream>>>(x_proj_w, w_x, n2);
        int n3 = DMOD * DI;        // out_proj_w
        k_f32_to_f16<<<(n3 + 255) / 256, 256, 0, stream>>>(out_proj_w, w_out, n3);
    }

    // 1) token conv + PE -> emb f16 [BL, 256]
    k_embed<<<BL, DMOD, 0, stream>>>(x_enc, tok_conv, emb);

    // 2) in_proj: xz = emb @ in_proj_w^T  (M=16384,N=1024,K=256), f16 out, 32x64 blocks
    {
        int total = (BL / 32) * (1024 / 64);     // 8192 wave-blocks
        k_gemm_nt_wmma_blk<2, 4, true><<<total / 4, 128, 0, stream>>>(emb, w_in, xz, BL, 1024, DMOD);
    }

    // 3) causal depthwise conv + SiLU -> u f16 [BL, 512]
    k_dconv_silu<<<BL, DI, 0, stream>>>(xz, dconv_w, dconv_b, u);

    // 4) x_proj: xdbl = u @ x_proj_w^T  (M=16384,N=48,K=512), f32 out, 16x48 blocks
    {
        int total = (BL / 16) * (48 / 48);       // 1024 wave-blocks
        k_gemm_nt_wmma_blk<1, 3, false><<<total / 4, 128, 0, stream>>>(u, w_x, xdbl, BL, 48, DI);
    }

    // 5) dt = softplus(dt_low @ dt_proj_w^T + b) -> f16 [BL, 512]
    k_dt_softplus<<<BL, DI, 0, stream>>>(xdbl, dt_proj_w, dt_proj_b, dtb);

    // 6) selective scan fused with skip/gate -> yg f16 [BL, 512]; 256 independent waves
    k_scan<<<dim3(BB, DI / 32), 32, 0, stream>>>(dtb, u, xz, xdbl, A_log, Dvec, yg);

    // 7) out_proj: mo = yg @ out_proj_w^T  (M=16384,N=256,K=512), f32 out, 32x64 blocks
    {
        int total = (BL / 32) * (DMOD / 64);     // 2048 wave-blocks
        k_gemm_nt_wmma_blk<2, 4, false><<<total / 4, 128, 0, stream>>>(yg, w_out, mo, BL, DMOD, DI);
    }

    // 8) LayerNorm + SiLU -> ma f32 [BL, 256]
    k_ln_silu<<<BL, DMOD, 0, stream>>>(mo, ln_g, ln_b, ma);

    // 9) heads
    k_head1<<<BL / 4, 128, 0, stream>>>(ma, cls_w, attn_w, attn_b, x_mark, sbuf, lgbuf);
    k_head2<<<BB, 256, 0, stream>>>(sbuf, lgbuf, out);
}
